// SparseTopKAttention_20186346291284
// MI455X (gfx1250) — matrix operand order repaired and checked by reading, not compiled, not run on hardware
//
#include <hip/hip_runtime.h>

typedef __attribute__((ext_vector_type(16))) __bf16 v16bf;
typedef __attribute__((ext_vector_type(8)))  float  v8f;

#define D_MODEL 256
#define NHEAD   8
#define DH      32
#define KSEL    32
#define NTOK    2048
#define NB      4
#define M_TOT   (NB * NTOK)   // 8192 tokens total

// ---------------------------------------------------------------------------
// Pack a 256x256 row-major fp32 weight matrix into bf16 WMMA B-fragments.
// Fragment layout (16x16x32 bf16 WMMA, B is 32x16):
//   lanes 0-15 : column = lane,    K = kt*32 + 0..15
//   lanes 16-31: column = lane-16, K = kt*32 + 16..31
// Stored as Wf[((kt*16 + nt)*32 + lane)*16 + j], 16 contiguous bf16 per lane.
// ---------------------------------------------------------------------------
__global__ void prep_w_kernel(const float* __restrict__ W, __bf16* __restrict__ Wf) {
    int t = blockIdx.x * blockDim.x + threadIdx.x;
    if (t >= 8 * 16 * 32) return;
    int lane = t & 31;
    int nt   = (t >> 5) & 15;
    int kt   = t >> 9;
    __bf16* dst = Wf + (((kt * 16 + nt) * 32 + lane) * 16);
    int col   = nt * 16 + (lane & 15);
    int kbase = kt * 32 + ((lane < 16) ? 0 : 16);
    #pragma unroll
    for (int j = 0; j < 16; ++j)
        dst[j] = (__bf16)W[(kbase + j) * D_MODEL + col];
}

// ---------------------------------------------------------------------------
// Y[M x 256] = X[M x 256] @ W[256 x 256] + bias, via v_wmma_f32_16x16x32_bf16.
// Block = 256 threads = 8 waves; blockIdx.y selects a 64-wide N band.
// The block's B fragments (8 kt x 4 nt x 32 lanes x 32B = 32KB) are staged
// into LDS once via async global->LDS copies (ASYNCcnt), then each wave
// computes a 32(M) x 64(N) tile: 2 A fragments x 4 B fragments per K-step,
// so every LDS B fragment feeds two WMMAs.
// Grid = (M/256, 4).
// ---------------------------------------------------------------------------
__global__ __launch_bounds__(256) void gemm_bias_kernel(
    const float* __restrict__ X, const __bf16* __restrict__ Wf,
    const float* __restrict__ bias, float* __restrict__ Y, int M) {
    __shared__ __bf16 Bsm[8 * 4 * 32 * 16];   // 32 KB
    int tid  = threadIdx.x;
    int wave = tid >> 5;
    int lane = tid & 31;
    int nt0  = blockIdx.y * 4;

    // ---- async stage: global Wf -> LDS, 16B chunk per lane per issue ----
    {
        const char* wbase = (const char*)Wf;
        unsigned lds0 = (unsigned)(size_t)(char*)Bsm;  // low 32 bits = LDS offset
        #pragma unroll
        for (int t = 0; t < 8; ++t) {
            int c    = tid + t * 256;        // 16-byte chunk id, 0..2047
            int e    = c >> 1;               // fragment-lane entry (32B each)
            int half = c & 1;
            int kt   = e >> 7;
            int rem  = e & 127;
            int n    = rem >> 5;
            int ln   = rem & 31;
            const char* gp = wbase + ((size_t)(((kt * 16) + nt0 + n) * 32 + ln) * 32
                                      + half * 16);
            unsigned la = lds0 + (unsigned)c * 16u;
            asm volatile("global_load_async_to_lds_b128 %0, %1, off"
                         :: "v"(la), "v"(gp) : "memory");
        }
        asm volatile("s_wait_asynccnt 0" ::: "memory");
    }
    __syncthreads();

    int mt0 = (blockIdx.x * 8 + wave) * 2;   // two consecutive 16-row M tiles
    if (mt0 * 16 >= M) return;
    int row0 = mt0 * 16 + (lane & 15);
    int row1 = row0 + 16;
    int kofs = (lane < 16) ? 0 : 16;         // A-fragment lane K offset

    v8f acc0[4] = {};
    v8f acc1[4] = {};
    #pragma unroll
    for (int kt = 0; kt < 8; ++kt) {
        const float* xp0 = X + (size_t)row0 * D_MODEL + kt * 32 + kofs;
        const float* xp1 = X + (size_t)row1 * D_MODEL + kt * 32 + kofs;
        v16bf a0, a1;
        #pragma unroll
        for (int j = 0; j < 8; ++j) {
            a0[j]     = (__bf16)xp0[j];
            a0[8 + j] = (__bf16)xp0[8 + j];
            a1[j]     = (__bf16)xp1[j];
            a1[8 + j] = (__bf16)xp1[8 + j];
        }
        #pragma unroll
        for (int n = 0; n < 4; ++n) {
            const __bf16* bp = Bsm + (((kt * 4 + n) * 32 + lane) << 4);
            v16bf bf = *(const v16bf*)bp;
            acc0[n] = __builtin_amdgcn_wmma_f32_16x16x32_bf16(
                false, a0, false, bf, (short)0, acc0[n], false, false);
            acc1[n] = __builtin_amdgcn_wmma_f32_16x16x32_bf16(
                false, a1, false, bf, (short)0, acc1[n], false, false);
        }
    }
    // epilogue: lane<16 -> rows 0..7 col lane; lane>=16 -> rows 8..15 col lane-16
    int mrow = (lane < 16) ? 0 : 8;
    int col  = lane & 15;
    #pragma unroll
    for (int n = 0; n < 4; ++n) {
        int ncol = (nt0 + n) * 16 + col;
        float bv = bias[ncol];
        #pragma unroll
        for (int r = 0; r < 8; ++r) {
            Y[(size_t)(mt0 * 16 + mrow + r) * D_MODEL + ncol]      = acc0[n][r] + bv;
            Y[(size_t)(mt0 * 16 + 16 + mrow + r) * D_MODEL + ncol] = acc1[n][r] + bv;
        }
    }
}

// ---------------------------------------------------------------------------
// In-place per-token, per-head rotation. Even heads: 2D positional RoPE
// (theta = [pos0*inv8, pos1*inv8], inv8[j]=BASE^(-j/8)); odd heads: heading
// dRoPE (theta[i] = heading*BASE^(-i/16)). One thread per (token, head, i<16).
// ---------------------------------------------------------------------------
__global__ void rope_kernel(float* __restrict__ Xp, const float* __restrict__ pos,
                            const float* __restrict__ heading) {
    int t = blockIdx.x * blockDim.x + threadIdx.x;
    if (t >= M_TOT * NHEAD * 16) return;
    int i   = t & 15;
    int h   = (t >> 4) & 7;
    int tok = t >> 7;
    float theta;
    if ((h & 1) == 0) {
        float p = (i < 8) ? pos[tok * 2 + 0] : pos[tok * 2 + 1];
        int j = i & 7;
        theta = p * powf(10000.0f, -(float)j / 8.0f);
    } else {
        theta = heading[tok] * powf(10000.0f, -(float)i / 16.0f);
    }
    float s, c;
    sincosf(theta, &s, &c);
    float* base = Xp + (size_t)tok * D_MODEL + h * DH;
    float x1 = base[i], x2 = base[16 + i];
    base[i]      = x1 * c - x2 * s;
    base[16 + i] = x1 * s + x2 * c;
}

// ---------------------------------------------------------------------------
// Top-32 nearest keys per query (squared 2D distance, mask -> +inf).
// One wave per query, 4 queries per 128-thread block.
// Distances staged in LDS; 32 rounds of scan + shfl_xor min-reduction
// with (dist, idx) tie-breaking (lowest index wins on ties).
// ---------------------------------------------------------------------------
__global__ __launch_bounds__(128) void topk_kernel(
    const float* __restrict__ pos_q, const float* __restrict__ pos_k,
    const unsigned char* __restrict__ mask_k, int* __restrict__ idx_out) {
    __shared__ float kx[NTOK];
    __shared__ float ky[NTOK];
    __shared__ float dist[4][NTOK];
    int wave = threadIdx.x >> 5;
    int lane = threadIdx.x & 31;
    int qg = blockIdx.x * 4 + wave;   // global query index [0, 8192)
    int b  = qg >> 11;                // batch (2048 queries per batch, aligned)

    for (int j = threadIdx.x; j < NTOK; j += blockDim.x) {
        kx[j] = pos_k[((size_t)b * NTOK + j) * 2 + 0];
        ky[j] = pos_k[((size_t)b * NTOK + j) * 2 + 1];
    }
    __syncthreads();

    float qx = pos_q[(size_t)qg * 2 + 0];
    float qy = pos_q[(size_t)qg * 2 + 1];
    for (int t = 0; t < 64; ++t) {
        int j = lane + t * 32;
        float dx = qx - kx[j], dy = qy - ky[j];
        float d2 = dx * dx + dy * dy;
        if (!mask_k[(size_t)b * NTOK + j]) d2 = __builtin_inff();
        dist[wave][j] = d2;
    }
    __syncthreads();

    for (int r = 0; r < KSEL; ++r) {
        float best = __builtin_inff();
        int   bidx = NTOK;
        for (int t = 0; t < 64; ++t) {
            int j = lane + t * 32;
            float d = dist[wave][j];
            if (d < best || (d == best && j < bidx)) { best = d; bidx = j; }
        }
        #pragma unroll
        for (int off = 16; off >= 1; off >>= 1) {
            float od = __shfl_xor(best, off, 32);
            int   oi = __shfl_xor(bidx, off, 32);
            if (od < best || (od == best && oi < bidx)) { best = od; bidx = oi; }
        }
        if (lane == 0) idx_out[(size_t)qg * KSEL + r] = bidx;
        if (lane == (bidx & 31)) dist[wave][bidx] = __builtin_inff();
        __syncthreads();
    }
}

// ---------------------------------------------------------------------------
// Gather attention: one wave per query, lane = selected key slot.
// logits over 8 heads x 32 dims; softmax across lanes per head; weighted V
// sum with lane owning head=lane/4, dims (lane%4)*8..+7 (contiguous 32B).
// ---------------------------------------------------------------------------
__global__ __launch_bounds__(256) void attn_kernel(
    const float* __restrict__ Qr, const float* __restrict__ Kr,
    const float* __restrict__ V, const int* __restrict__ idx,
    const unsigned char* __restrict__ mask_k, float* __restrict__ out) {
    __shared__ float wsh[8][32][NHEAD];
    int wave = threadIdx.x >> 5;
    int lane = threadIdx.x & 31;
    int qg = blockIdx.x * 8 + wave;
    int b  = qg >> 11;
    int ki = idx[(size_t)qg * KSEL + lane];

    const float* qrow = Qr + (size_t)qg * D_MODEL;
    const float* krow = Kr + ((size_t)b * NTOK + ki) * D_MODEL;
    const float scale = 0.17677669529663687f;  // 1/sqrt(32)
    bool valid = mask_k[(size_t)b * NTOK + ki] != 0;

    float logit[NHEAD];
    #pragma unroll
    for (int h = 0; h < NHEAD; ++h) {
        float acc = 0.f;
        #pragma unroll
        for (int d = 0; d < DH; ++d)
            acc += qrow[h * DH + d] * krow[h * DH + d];
        logit[h] = valid ? acc * scale : -__builtin_inff();
    }
    #pragma unroll
    for (int h = 0; h < NHEAD; ++h) {
        float m = logit[h];
        #pragma unroll
        for (int off = 16; off >= 1; off >>= 1)
            m = fmaxf(m, __shfl_xor(m, off, 32));
        float e = (logit[h] == -__builtin_inff()) ? 0.f : expf(logit[h] - m);
        float sum = e;
        #pragma unroll
        for (int off = 16; off >= 1; off >>= 1)
            sum += __shfl_xor(sum, off, 32);
        wsh[wave][lane][h] = e / sum;
    }
    __syncthreads();

    int hh    = lane >> 2;
    int dbase = hh * DH + (lane & 3) * 8;
    float acc[8] = {0.f, 0.f, 0.f, 0.f, 0.f, 0.f, 0.f, 0.f};
    for (int k = 0; k < KSEL; ++k) {
        float wk  = wsh[wave][k][hh];
        int  kidx = __shfl(ki, k, 32);
        const float* vrow = V + ((size_t)b * NTOK + kidx) * D_MODEL + dbase;
        #pragma unroll
        for (int j = 0; j < 8; ++j) acc[j] += wk * vrow[j];
    }
    float* orow = out + (size_t)qg * D_MODEL + dbase;
    #pragma unroll
    for (int j = 0; j < 8; ++j) orow[j] = acc[j];
}

// ---------------------------------------------------------------------------
extern "C" void kernel_launch(void* const* d_in, const int* in_sizes, int n_in,
                              void* d_out, int out_size, void* d_ws, size_t ws_size,
                              hipStream_t stream) {
    const float* q_feat    = (const float*)d_in[0];
    const float* kv_feat   = (const float*)d_in[1];
    const float* pos_q     = (const float*)d_in[2];
    const float* pos_k     = (const float*)d_in[3];
    const float* heading_q = (const float*)d_in[4];
    const float* heading_k = (const float*)d_in[5];
    const unsigned char* mask_k = (const unsigned char*)d_in[6];
    const float* Wq = (const float*)d_in[7];
    const float* bq = (const float*)d_in[8];
    const float* Wk = (const float*)d_in[9];
    const float* bk = (const float*)d_in[10];
    const float* Wv = (const float*)d_in[11];
    const float* bv = (const float*)d_in[12];
    const float* Wo = (const float*)d_in[13];
    const float* bo = (const float*)d_in[14];
    float* outp = (float*)d_out;

    // workspace carve-up
    char* ws = (char*)d_ws;
    const size_t MAT = (size_t)M_TOT * D_MODEL * sizeof(float);   // 8 MB
    float* Qr   = (float*)(ws);
    float* Krm  = (float*)(ws + MAT);
    float* Vv   = (float*)(ws + 2 * MAT);
    float* attn = (float*)(ws + 3 * MAT);
    const size_t WFRAG = (size_t)D_MODEL * D_MODEL * sizeof(__bf16); // 128 KB
    __bf16* WfQ = (__bf16*)(ws + 4 * MAT);
    __bf16* WfK = (__bf16*)(ws + 4 * MAT + WFRAG);
    __bf16* WfV = (__bf16*)(ws + 4 * MAT + 2 * WFRAG);
    __bf16* WfO = (__bf16*)(ws + 4 * MAT + 3 * WFRAG);
    int* idxb   = (int*)(ws + 4 * MAT + 4 * WFRAG);

    // 1. pack weights into WMMA B fragments (bf16)
    prep_w_kernel<<<16, 256, 0, stream>>>(Wq, WfQ);
    prep_w_kernel<<<16, 256, 0, stream>>>(Wk, WfK);
    prep_w_kernel<<<16, 256, 0, stream>>>(Wv, WfV);
    prep_w_kernel<<<16, 256, 0, stream>>>(Wo, WfO);

    // 2. Q/K/V projections (project-then-gather is exact for row-wise gathers)
    dim3 ggrid(M_TOT / 256, 4);
    gemm_bias_kernel<<<ggrid, 256, 0, stream>>>(q_feat,  WfQ, bq, Qr,  M_TOT);
    gemm_bias_kernel<<<ggrid, 256, 0, stream>>>(kv_feat, WfK, bk, Krm, M_TOT);
    gemm_bias_kernel<<<ggrid, 256, 0, stream>>>(kv_feat, WfV, bv, Vv,  M_TOT);

    // 3. per-token rotations (K rotation depends only on the key's pos/heading)
    int rope_threads = M_TOT * NHEAD * 16;
    rope_kernel<<<(rope_threads + 255) / 256, 256, 0, stream>>>(Qr,  pos_q, heading_q);
    rope_kernel<<<(rope_threads + 255) / 256, 256, 0, stream>>>(Krm, pos_k, heading_k);

    // 4. top-32 nearest-key selection
    topk_kernel<<<M_TOT / 4, 128, 0, stream>>>(pos_q, pos_k, mask_k, idxb);

    // 5. gather attention
    attn_kernel<<<M_TOT / 8, 256, 0, stream>>>(Qr, Krm, Vv, idxb, mask_k, attn);

    // 6. output projection straight into d_out
    gemm_bias_kernel<<<ggrid, 256, 0, stream>>>(attn, WfO, bo, outp, M_TOT);
}
